// NeuralODE_70574902608429
// MI455X (gfx1250) — matrix operand order repaired and checked by reading, not compiled, not run on hardware
//
#include <hip/hip_runtime.h>

typedef __attribute__((ext_vector_type(16))) _Float16 v16h;
typedef __attribute__((ext_vector_type(8)))  float    v8f;
typedef __attribute__((ext_vector_type(2)))  float    v2f;
typedef __attribute__((ext_vector_type(8)))  int      v8i;

// ---- Tsit5 coefficients ----
#define CA21 0.161f
#define CA31 (-0.008480655492356989f)
#define CA32 0.335480655492357f
#define CA41 2.8971530571054935f
#define CA42 (-6.359448489975075f)
#define CA43 4.3622954328695815f
#define CA51 5.325864828439257f
#define CA52 (-11.748883564062828f)
#define CA53 7.4955393428898365f
#define CA54 (-0.09249506636175525f)
#define CA61 5.86145544294642f
#define CA62 (-12.92096931784711f)
#define CA63 8.159367898576159f
#define CA64 (-0.071584973281401f)
#define CA65 (-0.028269050394068383f)
#define CB1 0.09646076681806523f
#define CB2 0.01f
#define CB3 0.4798896504144996f
#define CB4 1.379008574103742f
#define CB5 (-3.290069515436081f)
#define CB6 2.324710524099774f

struct F3 { float x, y, z; };

__device__ __forceinline__ float fast_tanh(float x) {
#if __has_builtin(__builtin_amdgcn_tanhf)
    return __builtin_amdgcn_tanhf(x);           // v_tanh_f32 (TRANS, 1 op)
#elif __has_builtin(__builtin_amdgcn_tanh_f32)
    return __builtin_amdgcn_tanh_f32(x);
#else
    // tanh(x) = sign(x) * (1 - 2/(exp(2|x|)+1)); exp via v_exp_f32 (2^t)
    float ax = __builtin_fabsf(x);
    float e  = __builtin_amdgcn_exp2f(ax * 2.8853900817779268f); // 2*log2(e)
    float r  = 1.0f - 2.0f * __builtin_amdgcn_rcpf(e + 1.0f);
    return __builtin_copysignf(r, x);
#endif
}

#if __has_builtin(__builtin_amdgcn_permlane16_swap)
#define HAVE_PLSWAP 1
#else
#define HAVE_PLSWAP 0
#endif

__device__ __forceinline__ unsigned swap16(unsigned v) {
    // ds_swizzle SWAPX16: group-of-32, xor_mask=0x10, and_mask=0x1f
    return (unsigned)__builtin_amdgcn_ds_swizzle((int)v, 0x401F);
}

// Given a (low-half data) and b (high-half data), produce
// lo = {a.lanes0-15, b.lanes0-15}, hi = {a.lanes16-31, b.lanes16-31}.
__device__ __forceinline__ void half_merge(unsigned a, unsigned b, bool hi_lane,
                                           unsigned& lo, unsigned& hi_out) {
#if HAVE_PLSWAP
    auto r = __builtin_amdgcn_permlane16_swap(a, b, false, false);
    lo = (unsigned)r[0];
    hi_out = (unsigned)r[1];
#else
    unsigned sa = swap16(a), sb = swap16(b);
    lo = hi_lane ? sb : a;
    hi_out = hi_lane ? b : sa;
#endif
}

// Broadcast low-half lanes' value to both halves.
__device__ __forceinline__ float bcast_lo(float v, bool hi_lane) {
#if HAVE_PLSWAP
    unsigned u = __builtin_bit_cast(unsigned, v);
    auto r = __builtin_amdgcn_permlane16_swap(u, u, false, false);
    return __builtin_bit_cast(float, (unsigned)r[0]);   // {v.lo, v.lo}
#else
    unsigned s = swap16(__builtin_bit_cast(unsigned, v));
    return hi_lane ? __builtin_bit_cast(float, s) : v;
#endif
}

__device__ __forceinline__ unsigned pkh2(float a, float b) {
    auto t = __builtin_amdgcn_cvt_pkrtz(a, b); // v_cvt_pk_rtz_f16_f32
    return __builtin_bit_cast(unsigned, t);
}

__global__ __launch_bounds__(256)
void tsit5_wmma_kernel(const float* __restrict__ ts, const float* __restrict__ y0,
                       const float* __restrict__ W1, const float* __restrict__ b1,
                       const float* __restrict__ W2, const float* __restrict__ b2,
                       const float* __restrict__ W3, const float* __restrict__ b3,
                       float* __restrict__ out, int T, int B)
{
    const int  lane  = (int)(threadIdx.x & 31u);
    const int  gwave = (int)((blockIdx.x * blockDim.x + threadIdx.x) >> 5);
    const int  lm    = lane & 15;
    const bool hi    = lane >= 16;
    const int  batch = gwave * 16 + lm;
    if (gwave * 16 >= B) return;   // wave-uniform exit keeps EXEC full

    // ---------- preload weight fragments (transposed formulation) ----------
    // Layer1: D1 = W1^T * [x y z 1]^T  via f32 WMMA 16x16x4, bias folded in K=3.
    v2f A1f[4];
    for (int t = 0; t < 4; ++t) {
        int c = 16 * t + lm;
        A1f[t][0] = hi ? W1[2 * 64 + c] : W1[0 * 64 + c];
        A1f[t][1] = hi ? b1[c]          : W1[1 * 64 + c];
    }
    // Layer2: A = W2^T tiles (4 M-tiles x 2 K-tiles), f16 16x16x32 fragments.
    v16h A2f[4][2];
    for (int tm = 0; tm < 4; ++tm)
        for (int kt = 0; kt < 2; ++kt) {
            int col = 16 * tm + lm;
            v16h a;
            for (int j = 0; j < 8; ++j) {
                int k0 = 32 * kt + (hi ? 16 : 0) + j;
                int k1 = 32 * kt + 8 + (hi ? 16 : 0) + j;
                a[j]     = (_Float16)W2[k0 * 64 + col];
                a[8 + j] = (_Float16)W2[k1 * 64 + col];
            }
            A2f[tm][kt] = a;
        }
    // Layer3: A = W3^T (M padded 3->16), 2 K-tiles.
    v16h A3f[2];
    for (int kt = 0; kt < 2; ++kt) {
        v16h a;
        for (int j = 0; j < 8; ++j) {
            int k0 = 32 * kt + (hi ? 16 : 0) + j;
            int k1 = 32 * kt + 8 + (hi ? 16 : 0) + j;
            a[j]     = (lm < 3) ? (_Float16)W3[k0 * 3 + lm] : (_Float16)0.0f;
            a[8 + j] = (lm < 3) ? (_Float16)W3[k1 * 3 + lm] : (_Float16)0.0f;
        }
        A3f[kt] = a;
    }
    // Layer2 bias as accumulator init: element (M,N) = b2[16*tm + M], M = r + 8*hi.
    v8f Cb2[4];
    for (int tm = 0; tm < 4; ++tm)
        for (int r = 0; r < 8; ++r)
            Cb2[tm][r] = b2[16 * tm + (hi ? 8 : 0) + r];
    const float b3x = b3[0], b3y = b3[1], b3z = b3[2];

    // ---------- state ----------
    F3 y;
    y.x = y0[batch * 3 + 0]; y.y = y0[batch * 3 + 1]; y.z = y0[batch * 3 + 2];
    if (!hi) {
        size_t o = (size_t)batch * 3;
        out[o + 0] = y.x; out[o + 1] = y.y; out[o + 2] = y.z;
    }

    // tanh + f16-pack + half-swap: C-layout D tiles -> B fragments for next layer
    auto buildB = [&](const v8f* D, v16h& B0, v16h& B1) {
        unsigned p[4][4];
        for (int t = 0; t < 4; ++t) {
            float th[8];
            for (int r = 0; r < 8; ++r) th[r] = fast_tanh(D[t][r]);
            for (int i = 0; i < 4; ++i) p[t][i] = pkh2(th[2 * i], th[2 * i + 1]);
        }
        v8i q0, q1;
        for (int i = 0; i < 4; ++i) {
            unsigned lo0, hi0, lo1, hi1;
            half_merge(p[0][i], p[1][i], hi, lo0, hi0);
            half_merge(p[2][i], p[3][i], hi, lo1, hi1);
            q0[i] = (int)lo0; q0[4 + i] = (int)hi0;
            q1[i] = (int)lo1; q1[4 + i] = (int)hi1;
        }
        B0 = __builtin_bit_cast(v16h, q0);
        B1 = __builtin_bit_cast(v16h, q1);
    };

    auto vf = [&](F3 in) -> F3 {
        const v8f zero = {0, 0, 0, 0, 0, 0, 0, 0};
        // B operand for layer1: per-lane homogeneous input (x,y,z,1)
        v2f By; By[0] = hi ? in.z : in.x; By[1] = hi ? 1.0f : in.y;
        v8f D1[4];
        for (int t = 0; t < 4; ++t)
            D1[t] = __builtin_amdgcn_wmma_f32_16x16x4_f32(
                false, A1f[t], false, By, (short)0, zero, false, false);
        v16h B0, B1;
        buildB(D1, B0, B1);

        v8f D2[4];
        for (int tm = 0; tm < 4; ++tm) {
            v8f acc = Cb2[tm];
            acc = __builtin_amdgcn_wmma_f32_16x16x32_f16(
                false, A2f[tm][0], false, B0, (short)0, acc, false, false);
            acc = __builtin_amdgcn_wmma_f32_16x16x32_f16(
                false, A2f[tm][1], false, B1, (short)0, acc, false, false);
            D2[tm] = acc;
        }
        v16h H0, H1;
        buildB(D2, H0, H1);

        v8f D3 = __builtin_amdgcn_wmma_f32_16x16x32_f16(
            false, A3f[0], false, H0, (short)0, zero, false, false);
        D3 = __builtin_amdgcn_wmma_f32_16x16x32_f16(
            false, A3f[1], false, H1, (short)0, D3, false, false);

        F3 r;
        r.x = bcast_lo(D3[0] + b3x, hi);
        r.y = bcast_lo(D3[1] + b3y, hi);
        r.z = bcast_lo(D3[2] + b3z, hi);
        return r;
    };

#define TRI(dst, E) do { dst.x = E(x); dst.y = E(y); dst.z = E(z); } while (0)

#pragma unroll 1
    for (int ti = 0; ti + 1 < T; ++ti) {
        float hstep = (ts[ti + 1] - ts[ti]) * 0.25f;
#pragma unroll 1
        for (int sidx = 0; sidx < 4; ++sidx) {
            F3 k1 = vf(y);
            F3 t2;
#define E2(c) (y.c + hstep * (CA21 * k1.c))
            TRI(t2, E2);
            F3 k2 = vf(t2);
            F3 t3;
#define E3(c) (y.c + hstep * (CA31 * k1.c + CA32 * k2.c))
            TRI(t3, E3);
            F3 k3 = vf(t3);
            F3 t4;
#define E4(c) (y.c + hstep * (CA41 * k1.c + CA42 * k2.c + CA43 * k3.c))
            TRI(t4, E4);
            F3 k4 = vf(t4);
            F3 t5;
#define E5(c) (y.c + hstep * (CA51 * k1.c + CA52 * k2.c + CA53 * k3.c + CA54 * k4.c))
            TRI(t5, E5);
            F3 k5 = vf(t5);
            F3 t6;
#define E6(c) (y.c + hstep * (CA61 * k1.c + CA62 * k2.c + CA63 * k3.c + CA64 * k4.c + CA65 * k5.c))
            TRI(t6, E6);
            F3 k6 = vf(t6);
#define EY(c) (y.c + hstep * (CB1 * k1.c + CB2 * k2.c + CB3 * k3.c + CB4 * k4.c + CB5 * k5.c + CB6 * k6.c))
            F3 yn;
            TRI(yn, EY);
            y = yn;
        }
        if (!hi) {
            size_t o = ((size_t)(ti + 1) * (size_t)B + (size_t)batch) * 3;
            out[o + 0] = y.x; out[o + 1] = y.y; out[o + 2] = y.z;
        }
    }
}

extern "C" void kernel_launch(void* const* d_in, const int* in_sizes, int n_in,
                              void* d_out, int out_size, void* d_ws, size_t ws_size,
                              hipStream_t stream) {
    const float* ts = (const float*)d_in[0];
    const float* y0 = (const float*)d_in[1];
    const float* W1 = (const float*)d_in[2];
    const float* b1 = (const float*)d_in[3];
    const float* W2 = (const float*)d_in[4];
    const float* b2 = (const float*)d_in[5];
    const float* W3 = (const float*)d_in[6];
    const float* b3 = (const float*)d_in[7];
    float* out = (float*)d_out;
    (void)n_in; (void)out_size; (void)d_ws; (void)ws_size;

    int T = in_sizes[0];
    int B = in_sizes[1] / 3;
    int waves   = (B + 15) / 16;
    int threads = waves * 32;
    int block   = 256;
    int grid    = (threads + block - 1) / block;
    hipLaunchKernelGGL(tsit5_wmma_kernel, dim3(grid), dim3(block), 0, stream,
                       ts, y0, W1, b1, W2, b2, W3, b3, out, T, B);
}
